// DGCNN_15771119911611
// MI455X (gfx1250) — matrix operand order repaired and checked by reading, not compiled, not run on hardware
//
#include <hip/hip_runtime.h>

typedef __bf16 bf16;
typedef __attribute__((ext_vector_type(16))) __bf16 v16bf;
typedef __attribute__((ext_vector_type(8)))  __bf16 v8bf;
typedef __attribute__((ext_vector_type(8)))  float  v8f;

#define BATCH 8
#define NPTS  2048
#define KNN   20

// ---------------------------------------------------------------- prep kernels
__global__ void k_transpose_x(const float* __restrict__ x, float* __restrict__ xT) {
  int i = blockIdx.x * 256 + threadIdx.x;            // over B*9*N
  if (i >= BATCH * 9 * NPTS) return;
  int n = i % NPTS; int c = (i / NPTS) % 9; int b = i / (9 * NPTS);
  xT[((size_t)b * NPTS + n) * 9 + c] = x[i];
}

__global__ void k_convert_w(const float* __restrict__ W, bf16* __restrict__ Wp,
                            int O, int C, int Opad, int Kpad) {
  int i = blockIdx.x * 256 + threadIdx.x;
  if (i >= Opad * Kpad) return;
  int r = i / Kpad, c = i % Kpad;
  float v = (r < O && c < C) ? W[r * C + c] : 0.0f;
  Wp[i] = (bf16)v;
}

__global__ void k_bn_prep(const float* __restrict__ g, const float* __restrict__ b,
                          const float* __restrict__ m, const float* __restrict__ v,
                          float* __restrict__ scale, float* __restrict__ shift, int O) {
  int i = blockIdx.x * 256 + threadIdx.x;
  if (i >= O) return;
  float s = g[i] * rsqrtf(v[i] + 1e-5f);
  scale[i] = s;
  shift[i] = b[i] - m[i] * s;
}

// ---------------------------------------------------------------- kNN (top-20)
__global__ __launch_bounds__(256)
void k_knn(const float* __restrict__ pts, int* __restrict__ idx,
           int Cfeat, int ldp, int coff) {
  __shared__ float ctr[64];
  __shared__ float dist[NPTS];
  __shared__ float bestv[256];
  __shared__ int   besti[256];
  const int n = blockIdx.x, b = blockIdx.y, tid = threadIdx.x;
  const float* base = pts + (size_t)b * NPTS * ldp + coff;
  if (tid < Cfeat) ctr[tid] = base[(size_t)n * ldp + tid];
  __syncthreads();
  for (int m = tid; m < NPTS; m += 256) {
    const float* pm = base + (size_t)m * ldp;
    float s = 0.f;
    for (int c = 0; c < Cfeat; ++c) { float d = pm[c] - ctr[c]; s += d * d; }
    dist[m] = -s;                                   // top-k of -||xi-xj||^2
  }
  __syncthreads();
  for (int k = 0; k < KNN; ++k) {
    float bv = -3.0e38f; int bi = 0;
    for (int m = tid; m < NPTS; m += 256) {
      float d = dist[m];
      if (d > bv || (d == bv && m < bi)) { bv = d; bi = m; }
    }
    bestv[tid] = bv; besti[tid] = bi;
    __syncthreads();
    for (int s = 128; s > 0; s >>= 1) {
      if (tid < s) {
        float ov = bestv[tid + s]; int oi = besti[tid + s];
        if (ov > bestv[tid] || (ov == bestv[tid] && oi < besti[tid])) {
          bestv[tid] = ov; besti[tid] = oi;
        }
      }
      __syncthreads();
    }
    if (tid == 0) {
      idx[((size_t)b * NPTS + n) * KNN + k] = besti[0];
      dist[besti[0]] = -3.4e38f;
    }
    __syncthreads();
  }
}

// ------------------------------------------------- edge features [nbr-ctr,ctr]
__global__ void k_edge(const float* __restrict__ pts, const int* __restrict__ idx,
                       bf16* __restrict__ E, int C, int ldp, int ldE) {
  long long i = (long long)blockIdx.x * 256 + threadIdx.x;   // B*N*K*ldE
  long long total = (long long)BATCH * NPTS * KNN * ldE;
  if (i >= total) return;
  int col = (int)(i % ldE);
  long long m = i / ldE;
  int k = (int)(m % KNN);
  long long nb = m / KNN;
  int n = (int)(nb % NPTS);
  int b = (int)(nb / NPTS);
  float v = 0.f;
  if (col < 2 * C) {
    const float* pb = pts + (size_t)b * NPTS * ldp;
    int cc = (col < C) ? col : (col - C);
    float ctr = pb[(size_t)n * ldp + cc];
    if (col < C) {
      int j = idx[((size_t)b * NPTS + n) * KNN + k];
      v = pb[(size_t)j * ldp + cc] - ctr;
    } else {
      v = ctr;
    }
  }
  E[i] = (bf16)v;
}

// --------------------------------------------------------------- WMMA GEMM
// Y[o][m] = act(bn(sum_c A[o][c] * B[m][c]))   (A: Opad x Kc bf16 row-major,
// B: per-batch M x ldB bf16 row-major). 8 waves/WG; wave = 16(o) x 32(m) tile.
// K-panel of 64 per barrier: 4 v_wmma issues per wave per sync window.
template <int WO, int WM>
__global__ __launch_bounds__(256)
void k_gemm_wmma(const bf16* __restrict__ A, const bf16* __restrict__ Bmat,
                 const float* __restrict__ scale, const float* __restrict__ shift,
                 bf16* __restrict__ OutH, float* __restrict__ OutF,
                 int M, int Kc, int ldB, long long strideB,
                 int ldO, long long strideO, int Oreal, int flags) {
  constexpr int OT  = WO * 16;      // block O tile
  constexpr int MT  = WM * 32;      // block M tile
  constexpr int LDT = 72;           // 64 data + 8 pad bf16 (144B rows, 16B-mult)
  __shared__ bf16 smA[OT * LDT];
  __shared__ bf16 smB[MT * LDT];
  const int tid = threadIdx.x;
  const int wave = tid >> 5, lane = tid & 31;
  const int lo = lane & 15, hi = lane >> 4;
  const int oBlock = blockIdx.y * OT;
  const int mBlock = blockIdx.x * MT;
  const int b = blockIdx.z;
  const bf16* Bbase = Bmat + (size_t)b * strideB;
  const int wo = wave % WO, wm = wave / WO;
  const int oW = wo * 16;
  const int mW = wm * 32;
  v8f acc0 = {};
  v8f acc1 = {};
  for (int kc = 0; kc < Kc; kc += 64) {
    if (kc + 64 < Kc)                      // prefetch next K-panel into GL2
      __builtin_prefetch((const void*)&Bbase[(size_t)mBlock * ldB + kc + 64], 0, 1);
    __syncthreads();
#pragma unroll
    for (int i = tid * 8; i < OT * 64; i += 2048) {
      int r = i >> 6, c = i & 63;
      *(v8bf*)&smA[r * LDT + c] =
          *(const v8bf*)&A[(size_t)(oBlock + r) * Kc + kc + c];
    }
#pragma unroll
    for (int i = tid * 8; i < MT * 64; i += 2048) {
      int r = i >> 6, c = i & 63;
      *(v8bf*)&smB[r * LDT + c] =
          *(const v8bf*)&Bbase[(size_t)(mBlock + r) * ldB + kc + c];
    }
    __syncthreads();
#pragma unroll
    for (int p = 0; p < 2; ++p) {
      const int ko = p * 32;
      // A fragment (16x32; lanes 0-15: K0-7/K16-23, lanes 16-31: K8-15/K24-31)
      v8bf a0 = *(const v8bf*)&smA[(oW + lo) * LDT + ko + hi * 8];
      v8bf a1 = *(const v8bf*)&smA[(oW + lo) * LDT + ko + 16 + hi * 8];
      v16bf av = __builtin_shufflevector(a0, a1, 0,1,2,3,4,5,6,7,8,9,10,11,12,13,14,15);
      // B fragments (32x16; lane = output column, lanes 0-15: K0-7/K16-23, lanes 16-31: K8-15/K24-31)
      v8bf b00 = *(const v8bf*)&smB[(mW + lo) * LDT + ko + hi * 8];
      v8bf b01 = *(const v8bf*)&smB[(mW + lo) * LDT + ko + 16 + hi * 8];
      v16bf bv0 = __builtin_shufflevector(b00, b01, 0,1,2,3,4,5,6,7,8,9,10,11,12,13,14,15);
      v8bf b10 = *(const v8bf*)&smB[(mW + 16 + lo) * LDT + ko + hi * 8];
      v8bf b11 = *(const v8bf*)&smB[(mW + 16 + lo) * LDT + ko + 16 + hi * 8];
      v16bf bv1 = __builtin_shufflevector(b10, b11, 0,1,2,3,4,5,6,7,8,9,10,11,12,13,14,15);
      acc0 = __builtin_amdgcn_wmma_f32_16x16x32_bf16(false, av, false, bv0,
                                                     (short)0, acc0, false, false);
      acc1 = __builtin_amdgcn_wmma_f32_16x16x32_bf16(false, av, false, bv1,
                                                     (short)0, acc1, false, false);
    }
  }
  // epilogue: BN + LeakyReLU(0.2), store
  const int oBase = oBlock + oW + hi * 8;           // element e -> row oBase+e
#pragma unroll
  for (int t = 0; t < 2; ++t) {
    v8f acc = t ? acc1 : acc0;
    int mG = mBlock + mW + t * 16 + lo;             // output column
    if (flags & 1) {
#pragma unroll
      for (int e = 0; e < 8; ++e) {
        float y = acc[e] * scale[oBase + e] + shift[oBase + e];
        acc[e] = (y >= 0.f) ? y : 0.2f * y;
      }
    }
    if (flags & 2) {                                // final layer: f32 (B,Oreal,N)
#pragma unroll
      for (int e = 0; e < 8; ++e) {
        int o = oBase + e;
        if (o < Oreal) OutF[((size_t)b * Oreal + o) * M + mG] = acc[e];
      }
    } else {                                        // bf16 [m][o], contiguous 16B
      v8bf ov;
#pragma unroll
      for (int e = 0; e < 8; ++e) ov[e] = (bf16)acc[e];
      *(v8bf*)&OutH[(size_t)b * strideO + (size_t)mG * ldO + oBase] = ov;
    }
  }
}

// ------------------------------------------------------- max over k neighbors
__global__ void k_maxk(const bf16* __restrict__ H, float* __restrict__ xo,
                       bf16* __restrict__ comb, int combOff) {
  int i = blockIdx.x * 256 + threadIdx.x;           // B*N*64
  if (i >= BATCH * NPTS * 64) return;
  int c = i & 63; int nb = i >> 6; int n = nb % NPTS; int b = nb / NPTS;
  const bf16* h = H + ((size_t)(b * NPTS + n) * KNN) * 64 + c;
  float v = -3.4e38f;
  for (int k = 0; k < KNN; ++k) v = fmaxf(v, (float)h[(size_t)k * 64]);
  xo[i] = v;
  comb[((size_t)b * NPTS + n) * 192 + combOff + c] = (bf16)v;
}

// ---------------------------------------------------------- global max over N
__global__ void k_gmax(const bf16* __restrict__ h6, bf16* __restrict__ g) {
  int i = blockIdx.x * 256 + threadIdx.x;           // B*1024
  if (i >= BATCH * 1024) return;
  int o = i % 1024, b = i / 1024;
  const bf16* p = h6 + (size_t)b * NPTS * 1024 + o;
  float v = -3.4e38f;
  for (int n = 0; n < NPTS; ++n) v = fmaxf(v, (float)p[(size_t)n * 1024]);
  g[i] = (bf16)v;
}

// ---------------------------------------------------- concat [g | x1 x2 x3]
__global__ void k_cat(const bf16* __restrict__ g, const bf16* __restrict__ x123,
                      bf16* __restrict__ cat) {
  long long i = (long long)blockIdx.x * 256 + threadIdx.x;   // B*N*1216
  if (i >= (long long)BATCH * NPTS * 1216) return;
  int c = (int)(i % 1216);
  long long nb = i / 1216;
  int n = (int)(nb % NPTS); int b = (int)(nb / NPTS);
  cat[i] = (c < 1024) ? g[b * 1024 + c]
                      : x123[((size_t)b * NPTS + n) * 192 + (c - 1024)];
}

// ------------------------------------------------------------------- host
extern "C" void kernel_launch(void* const* d_in, const int* in_sizes, int n_in,
                              void* d_out, int out_size, void* d_ws, size_t ws_size,
                              hipStream_t stream) {
  (void)out_size; (void)ws_size;
  const float* x = nullptr;
  const float *W[9], *Gg[8], *Bt[8], *Mm[8], *Vv[8];
  if (n_in >= 42 && in_sizes[0] == BATCH * 9 * NPTS && in_sizes[2] == 64) {
    // dict-insertion order: x, (W,g,b,m,v)x8, W9
    x = (const float*)d_in[0];
    for (int i = 0; i < 8; ++i) {
      W[i]  = (const float*)d_in[1 + 5 * i];
      Gg[i] = (const float*)d_in[2 + 5 * i];
      Bt[i] = (const float*)d_in[3 + 5 * i];
      Mm[i] = (const float*)d_in[4 + 5 * i];
      Vv[i] = (const float*)d_in[5 + 5 * i];
    }
    W[8] = (const float*)d_in[41];
  } else if (n_in >= 42 && in_sizes[0] == BATCH * 9 * NPTS) {
    // x first, params alphabetical: W1..W9, b1..b8, g1..g8, m1..m8, v1..v8
    x = (const float*)d_in[0];
    for (int i = 0; i < 9; ++i) W[i] = (const float*)d_in[1 + i];
    for (int i = 0; i < 8; ++i) {
      Bt[i] = (const float*)d_in[10 + i];
      Gg[i] = (const float*)d_in[18 + i];
      Mm[i] = (const float*)d_in[26 + i];
      Vv[i] = (const float*)d_in[34 + i];
    }
  } else {
    // fully alphabetical: params first, x last
    for (int i = 0; i < 9; ++i) W[i] = (const float*)d_in[i];
    for (int i = 0; i < 8; ++i) {
      Bt[i] = (const float*)d_in[9 + i];
      Gg[i] = (const float*)d_in[17 + i];
      Mm[i] = (const float*)d_in[25 + i];
      Vv[i] = (const float*)d_in[33 + i];
    }
    x = (const float*)d_in[n_in - 1];
  }

  // ---- workspace bump allocator (256B aligned) ----
  char* ws = (char*)d_ws;
  size_t off = 0;
  auto alloc = [&](size_t bytes) -> char* {
    off = (off + 255) & ~(size_t)255;
    char* p = ws + off;
    off += bytes;
    return p;
  };
  float* xT   = (float*)alloc((size_t)BATCH * NPTS * 9 * 4);
  int*   idx  = (int*)  alloc((size_t)BATCH * NPTS * KNN * 4);
  float* x1   = (float*)alloc((size_t)BATCH * NPTS * 64 * 4);
  float* x2   = (float*)alloc((size_t)BATCH * NPTS * 64 * 4);
  float* x3   = (float*)alloc((size_t)BATCH * NPTS * 64 * 4);
  bf16*  x123 = (bf16*) alloc((size_t)BATCH * NPTS * 192 * 2);
  bf16*  gbuf = (bf16*) alloc((size_t)BATCH * 1024 * 2);
  float* scaleB = (float*)alloc(8 * 1024 * 4);
  float* shiftB = (float*)alloc(8 * 1024 * 4);
  static const int Od[9] = {64, 64, 64, 64, 64, 1024, 512, 256, 13};
  static const int Cd[9] = {18, 64, 128, 64, 128, 192, 1216, 512, 256};
  static const int Op[9] = {64, 64, 64, 64, 64, 1024, 512, 256, 16};
  static const int Kp[9] = {64, 64, 128, 64, 128, 192, 1216, 512, 256};
  bf16* Wp[9];
  for (int i = 0; i < 9; ++i) Wp[i] = (bf16*)alloc((size_t)Op[i] * Kp[i] * 2);
  const int ME = NPTS * KNN;                       // 40960 edge rows / batch
  size_t szE   = (size_t)BATCH * ME * 128 * 2;     // 84 MB
  size_t szCat = (size_t)BATCH * NPTS * 1216 * 2;  // 40 MB
  bf16* regA = (bf16*)alloc(szE > szCat ? szE : szCat);
  bf16* regB = (bf16*)alloc((size_t)BATCH * ME * 64 * 2);   // 42 MB
  bf16* H7 = regB;
  bf16* H8 = regB + (size_t)BATCH * NPTS * 512;

  auto cdiv = [](long long a, long long b) { return (int)((a + b - 1) / b); };

  // ---- prep: transpose x, convert weights to bf16, fold BN ----
  k_transpose_x<<<cdiv((long long)BATCH * 9 * NPTS, 256), 256, 0, stream>>>(x, xT);
  for (int i = 0; i < 9; ++i)
    k_convert_w<<<cdiv((long long)Op[i] * Kp[i], 256), 256, 0, stream>>>(
        W[i], Wp[i], Od[i], Cd[i], Op[i], Kp[i]);
  for (int i = 0; i < 8; ++i)
    k_bn_prep<<<cdiv(Od[i], 256), 256, 0, stream>>>(
        Gg[i], Bt[i], Mm[i], Vv[i], scaleB + i * 1024, shiftB + i * 1024, Od[i]);

  auto gemm = [&](const bf16* Aw, const bf16* Bm, int layer, bf16* OutH, float* OutF,
                  int M, int Opad, int Kc, int ldB, long long strideB,
                  int ldO, long long strideO, int Oreal, int flags) {
    const float* sc = scaleB + (layer < 8 ? layer : 0) * 1024;
    const float* sh = shiftB + (layer < 8 ? layer : 0) * 1024;
    if (Opad >= 64) {
      dim3 grid(M / 64, Opad / 64, BATCH);
      k_gemm_wmma<4, 2><<<grid, 256, 0, stream>>>(Aw, Bm, sc, sh, OutH, OutF, M, Kc,
                                                  ldB, strideB, ldO, strideO, Oreal, flags);
    } else {
      dim3 grid(M / 256, Opad / 16, BATCH);
      k_gemm_wmma<1, 8><<<grid, 256, 0, stream>>>(Aw, Bm, sc, sh, OutH, OutF, M, Kc,
                                                  ldB, strideB, ldO, strideO, Oreal, flags);
    }
  };

  // ---- block 1: knn on xyz (ch 6:9), edge(18 -> pad 64), conv1, conv2, maxk ----
  k_knn<<<dim3(NPTS, BATCH), 256, 0, stream>>>(xT, idx, 3, 9, 6);
  k_edge<<<cdiv((long long)BATCH * ME * 64, 256), 256, 0, stream>>>(xT, idx, regA, 9, 9, 64);
  gemm(Wp[0], regA, 0, regB, nullptr, ME, 64, 64, 64, (long long)ME * 64,
       64, (long long)ME * 64, 64, 1);
  gemm(Wp[1], regB, 1, regA, nullptr, ME, 64, 64, 64, (long long)ME * 64,
       64, (long long)ME * 64, 64, 1);
  k_maxk<<<cdiv((long long)BATCH * NPTS * 64, 256), 256, 0, stream>>>(regA, x1, x123, 0);

  // ---- block 2 ----
  k_knn<<<dim3(NPTS, BATCH), 256, 0, stream>>>(x1, idx, 64, 64, 0);
  k_edge<<<cdiv((long long)BATCH * ME * 128, 256), 256, 0, stream>>>(x1, idx, regA, 64, 64, 128);
  gemm(Wp[2], regA, 2, regB, nullptr, ME, 64, 128, 128, (long long)ME * 128,
       64, (long long)ME * 64, 64, 1);
  gemm(Wp[3], regB, 3, regA, nullptr, ME, 64, 64, 64, (long long)ME * 64,
       64, (long long)ME * 64, 64, 1);
  k_maxk<<<cdiv((long long)BATCH * NPTS * 64, 256), 256, 0, stream>>>(regA, x2, x123, 64);

  // ---- block 3 ----
  k_knn<<<dim3(NPTS, BATCH), 256, 0, stream>>>(x2, idx, 64, 64, 0);
  k_edge<<<cdiv((long long)BATCH * ME * 128, 256), 256, 0, stream>>>(x2, idx, regA, 64, 64, 128);
  gemm(Wp[4], regA, 4, regB, nullptr, ME, 64, 128, 128, (long long)ME * 128,
       64, (long long)ME * 64, 64, 1);
  k_maxk<<<cdiv((long long)BATCH * NPTS * 64, 256), 256, 0, stream>>>(regB, x3, x123, 128);

  // ---- global embedding + head ----
  gemm(Wp[5], x123, 5, regB, nullptr, NPTS, 1024, 192, 192, (long long)NPTS * 192,
       1024, (long long)NPTS * 1024, 1024, 1);                      // h6
  k_gmax<<<cdiv((long long)BATCH * 1024, 256), 256, 0, stream>>>(regB, gbuf);
  k_cat<<<cdiv((long long)BATCH * NPTS * 1216, 256), 256, 0, stream>>>(gbuf, x123, regA);
  gemm(Wp[6], regA, 6, H7, nullptr, NPTS, 512, 1216, 1216, (long long)NPTS * 1216,
       512, (long long)NPTS * 512, 512, 1);                         // conv7
  gemm(Wp[7], H7, 7, H8, nullptr, NPTS, 256, 512, 512, (long long)NPTS * 512,
       256, (long long)NPTS * 256, 256, 1);                         // conv8
  gemm(Wp[8], H8, 8, nullptr, (float*)d_out, NPTS, 16, 256, 256, (long long)NPTS * 256,
       16, 0, 13, 2);                                               // final 13-ch
}
